// DeltaRNN_36240934044045
// MI455X (gfx1250) — matrix operand order repaired and checked by reading, not compiled, not run on hardware
//
#include <hip/hip_runtime.h>

typedef __attribute__((ext_vector_type(16))) _Float16 v16h;
typedef __attribute__((ext_vector_type(8)))  _Float16 v8h;
typedef __attribute__((ext_vector_type(8)))  float    v8f;

#define T_STEPS 512
#define BATCH   64
#define ISZ     512
#define HSZ     1024
#define GSZ     4096            // 4*HSZ

// d_out layout (floats): [T,B,H] out | x_prev[B,1024] | h[B,H] | h_prev[B,H] | c[B,H] | m[B,G] | reg[1]
#define OFF_XPREV ((size_t)T_STEPS * BATCH * HSZ)
#define N_ZERO    ((size_t)BATCH * 1024 + 3ull * BATCH * HSZ)   // x_prev,h,h_prev,c (contiguous)
#define N_M       ((size_t)BATCH * GSZ)

__device__ __forceinline__ float quant_a(float x) {   // Q8.8, clip [-2^15, 2^15-1]
  float r = rintf(x * 256.0f);
  r = fminf(fmaxf(r, -32768.0f), 32767.0f);
  return r * 0.00390625f;
}
__device__ __forceinline__ float quant_n(float x) {   // Q2.8, clip [-512, 511]
  float r = rintf(x * 256.0f);
  r = fminf(fmaxf(r, -512.0f), 511.0f);
  return r * 0.00390625f;
}
__device__ __forceinline__ float sigm(float x) { return 1.0f / (1.0f + __expf(-x)); }
__device__ __forceinline__ float tanh_fast(float x) { return 2.0f * sigm(2.0f * x) - 1.0f; }

// ---------------------------------------------------------------------------
// One-time (per launch) init: zero state, fold biases into m, fp32->f16 weights
// ---------------------------------------------------------------------------
__global__ void __launch_bounds__(256)
delta_init_kernel(const float* __restrict__ wih, const float* __restrict__ whh,
                  const float* __restrict__ bih, const float* __restrict__ bhh,
                  float* __restrict__ state,          // d_out + OFF_XPREV
                  _Float16* __restrict__ wih16, _Float16* __restrict__ whh16) {
  const size_t NW1 = (size_t)GSZ * ISZ;
  const size_t NW2 = (size_t)GSZ * HSZ;
  const size_t total = N_ZERO + N_M + 1 + NW1 + NW2;
  for (size_t i = (size_t)blockIdx.x * blockDim.x + threadIdx.x; i < total;
       i += (size_t)gridDim.x * blockDim.x) {
    size_t r = i;
    if (r < N_ZERO) { state[r] = 0.0f; continue; }
    r -= N_ZERO;
    if (r < N_M) { state[N_ZERO + r] = bih[r % GSZ] + bhh[r % GSZ]; continue; }
    r -= N_M;
    if (r == 0) { state[N_ZERO + N_M] = 0.0f; continue; }   // reg
    r -= 1;
    if (r < NW1) { wih16[r] = (_Float16)wih[r]; continue; }
    r -= NW1;
    whh16[r] = (_Float16)whh[r];
  }
}

// ---------------------------------------------------------------------------
// Per-step delta: masked dx/dh (f16 for WMMA), prev-state update, reg L1 accum
// ---------------------------------------------------------------------------
__global__ void __launch_bounds__(256)
delta_mask_kernel(const float* __restrict__ xt, float* __restrict__ xprev,
                  const float* __restrict__ h, float* __restrict__ hprev,
                  _Float16* __restrict__ dx16, _Float16* __restrict__ dh16,
                  float* __restrict__ reg) {
  const float TH_Q = 0.1015625f;          // quantize(0.1, Q8.8) = 26/256
  const int tid = threadIdx.x;
  const int idx = blockIdx.x * 256 + tid; // grid exactly covers B*I + B*H
  const int N1 = BATCH * ISZ;
  float local = 0.0f;
  if (idx < N1) {
    int b = idx / ISZ, i = idx - b * ISZ;
    float xq = quant_a(xt[idx]);
    int pidx = b * 1024 + i;              // x_prev is padded [B,1024]
    float xp = xprev[pidx];
    float dx = xq - xp;
    float ad = fabsf(dx);
    dx16[idx] = (_Float16)((ad < TH_Q) ? 0.0f : dx);
    if (ad >= 0.1f) xprev[pidx] = xq;
  } else {
    int j = idx - N1;                     // j in [0, B*H)
    float hv = h[j], hp = hprev[j];
    float dh = hv - hp;
    float ad = fabsf(dh);
    float dhm = (ad < TH_Q) ? 0.0f : dh;
    dh16[j] = (_Float16)dhm;
    if (ad >= 0.1f) hprev[j] = hv;
    local = fabsf(dhm);
  }
  __shared__ float sred[256];
  sred[tid] = local;
  __syncthreads();
  for (int s = 128; s > 0; s >>= 1) {
    if (tid < s) sred[tid] += sred[tid + s];
    __syncthreads();
  }
  if (tid == 0) atomicAdd(reg, sred[0]);
}

// ---------------------------------------------------------------------------
// Per-step fused delta-GEMM (WMMA f16->f32) + quantized LSTM gate math.
// Block = 128 thr (4 waves, one per SIMD32). Wave w owns M-tile w and ALL
// four gates for the block's 16 H-columns: A fragment loaded once, reused by
// 4 WMMAs, and each lane ends up holding p_i/p_f/p_g/p_o for the same (b,j)
// in its four accumulators -> gate/cell/hidden math is pure in-register VALU,
// no LDS, no barrier.
// ---------------------------------------------------------------------------
__global__ void __launch_bounds__(128)
lstm_step_kernel(const _Float16* __restrict__ dx16, const _Float16* __restrict__ dh16,
                 const _Float16* __restrict__ wih16, const _Float16* __restrict__ whh16,
                 float* __restrict__ m, float* __restrict__ c,
                 float* __restrict__ h, float* __restrict__ out_t) {
  const int tid   = threadIdx.x;
  const int lane  = tid & 31;
  const int mtile = tid >> 5;             // 0..3
  const int n0    = blockIdx.x * 16;      // column tile within H

  // A fragment addressing (16-bit A 16x32, wave32): contiguous 16 K-values
  const int arow  = mtile * 16 + (lane & 15);
  const int akoff = (lane < 16) ? 0 : 16;
  // B fragment addressing (16-bit B 32x16, same k order as A): contiguous 16 K-values
  const int bn    = lane & 15;
  const int bkoff = (lane < 16) ? 0 : 16;

  v8f acc0 = {}, acc1 = {}, acc2 = {}, acc3 = {};   // gates i, f, g, o

  // ---- dx @ W_ih^T ----
  {
    const _Float16* ab = dx16 + (size_t)arow * ISZ + akoff;
    const _Float16* wb = wih16 + (size_t)(n0 + bn) * ISZ + bkoff;
    const size_t gstride = (size_t)HSZ * ISZ;       // one gate slice of W_ih
#pragma unroll 4
    for (int k0 = 0; k0 < ISZ; k0 += 32) {
      v8h alo = *(const v8h*)(ab + k0);
      v8h ahi = *(const v8h*)(ab + k0 + 8);
      v16h a = __builtin_shufflevector(alo, ahi, 0,1,2,3,4,5,6,7,8,9,10,11,12,13,14,15);
      v16h b0 = *(const v16h*)(wb + k0);
      v16h b1 = *(const v16h*)(wb + gstride + k0);
      v16h b2 = *(const v16h*)(wb + 2 * gstride + k0);
      v16h b3 = *(const v16h*)(wb + 3 * gstride + k0);
      __builtin_prefetch(wb + k0 + 128, 0, 1);
      acc0 = __builtin_amdgcn_wmma_f32_16x16x32_f16(false, a, false, b0, (short)0, acc0, false, false);
      acc1 = __builtin_amdgcn_wmma_f32_16x16x32_f16(false, a, false, b1, (short)0, acc1, false, false);
      acc2 = __builtin_amdgcn_wmma_f32_16x16x32_f16(false, a, false, b2, (short)0, acc2, false, false);
      acc3 = __builtin_amdgcn_wmma_f32_16x16x32_f16(false, a, false, b3, (short)0, acc3, false, false);
    }
  }
  // ---- dh @ W_hh^T ----
  {
    const _Float16* ab = dh16 + (size_t)arow * HSZ + akoff;
    const _Float16* wb = whh16 + (size_t)(n0 + bn) * HSZ + bkoff;
    const size_t gstride = (size_t)HSZ * HSZ;       // one gate slice of W_hh
#pragma unroll 4
    for (int k0 = 0; k0 < HSZ; k0 += 32) {
      v8h alo = *(const v8h*)(ab + k0);
      v8h ahi = *(const v8h*)(ab + k0 + 8);
      v16h a = __builtin_shufflevector(alo, ahi, 0,1,2,3,4,5,6,7,8,9,10,11,12,13,14,15);
      v16h b0 = *(const v16h*)(wb + k0);
      v16h b1 = *(const v16h*)(wb + gstride + k0);
      v16h b2 = *(const v16h*)(wb + 2 * gstride + k0);
      v16h b3 = *(const v16h*)(wb + 3 * gstride + k0);
      __builtin_prefetch(wb + k0 + 128, 0, 1);
      acc0 = __builtin_amdgcn_wmma_f32_16x16x32_f16(false, a, false, b0, (short)0, acc0, false, false);
      acc1 = __builtin_amdgcn_wmma_f32_16x16x32_f16(false, a, false, b1, (short)0, acc1, false, false);
      acc2 = __builtin_amdgcn_wmma_f32_16x16x32_f16(false, a, false, b2, (short)0, acc2, false, false);
      acc3 = __builtin_amdgcn_wmma_f32_16x16x32_f16(false, a, false, b3, (short)0, acc3, false, false);
    }
  }

  // ---- m accumulate + gates + cell + hidden, fully in registers ----
  // C/D layout: VGPR v -> row (v + 8*(lane>=16)), col = lane&15
  const int rbase = mtile * 16 + ((lane < 16) ? 0 : 8);
  const int ccol  = lane & 15;
#pragma unroll
  for (int v = 0; v < 8; ++v) {
    const int brow = rbase + v;
    const size_t mrow = (size_t)brow * GSZ + n0 + ccol;
    float mi = m[mrow]           + acc0[v];
    float mf = m[mrow +     HSZ] + acc1[v];
    float mg = m[mrow + 2 * HSZ] + acc2[v];
    float mo = m[mrow + 3 * HSZ] + acc3[v];
    m[mrow]           = mi;
    m[mrow +     HSZ] = mf;
    m[mrow + 2 * HSZ] = mg;
    m[mrow + 3 * HSZ] = mo;
    float gi = quant_n(sigm(quant_a(mi)));
    float gf = quant_n(sigm(quant_a(mf)));
    float gg = quant_n(tanh_fast(quant_a(mg)));
    float go = quant_n(sigm(quant_a(mo)));
    const size_t ci = (size_t)brow * HSZ + n0 + ccol;
    float qc = quant_a(c[ci] * gf + gi * gg);
    c[ci] = qc;
    float hn = quant_a(go * quant_n(tanh_fast(qc)));
    h[ci] = hn;
    out_t[ci] = hn;
  }
}

// ---------------------------------------------------------------------------
extern "C" void kernel_launch(void* const* d_in, const int* in_sizes, int n_in,
                              void* d_out, int out_size, void* d_ws, size_t ws_size,
                              hipStream_t stream) {
  const float* x   = (const float*)d_in[0];
  const float* wih = (const float*)d_in[1];
  const float* whh = (const float*)d_in[2];
  const float* bih = (const float*)d_in[3];
  const float* bhh = (const float*)d_in[4];

  float* out   = (float*)d_out;
  float* state = out + OFF_XPREV;                 // x_prev | h | h_prev | c | m | reg
  float* xprev = state;
  float* h     = state + (size_t)BATCH * 1024;
  float* hprev = h + (size_t)BATCH * HSZ;
  float* c     = hprev + (size_t)BATCH * HSZ;
  float* m     = c + (size_t)BATCH * HSZ;
  float* reg   = state + N_ZERO + N_M;

  _Float16* wih16 = (_Float16*)d_ws;
  _Float16* whh16 = wih16 + (size_t)GSZ * ISZ;
  _Float16* dx16  = whh16 + (size_t)GSZ * HSZ;
  _Float16* dh16  = dx16 + (size_t)BATCH * ISZ;

  delta_init_kernel<<<2048, 256, 0, stream>>>(wih, whh, bih, bhh, state, wih16, whh16);

  for (int t = 0; t < T_STEPS; ++t) {
    delta_mask_kernel<<<(BATCH * ISZ + BATCH * HSZ) / 256, 256, 0, stream>>>(
        x + (size_t)t * BATCH * ISZ, xprev, h, hprev, dx16, dh16, reg);
    lstm_step_kernel<<<HSZ / 16, 128, 0, stream>>>(
        dx16, dh16, wih16, whh16, m, c, h, out + (size_t)t * BATCH * HSZ);
  }
}
